// Attention_10196252361157
// MI455X (gfx1250) — matrix operand order repaired and checked by reading, not compiled, not run on hardware
//
#include <hip/hip_runtime.h>
#include <hip/hip_bf16.h>

typedef __attribute__((ext_vector_type(16))) __bf16 v16bf;
typedef __attribute__((ext_vector_type(8)))  __bf16 bf16x8;
typedef __attribute__((ext_vector_type(8)))  float  v8f;

#define B_  16
#define L_  8192
#define F_  512
#define I_  512
#define A_  256

#define LT   128   // L rows per workgroup (8 waves x 16)
#define KC   128   // K (feature) chunk staged in LDS
#define LDK  136   // padded K stride in LDS (bf16 elems): 272B row, 16B aligned,
                   // bank step 68 mod 64 = 4 -> conflict-free b128 frag reads

// ---------------------------------------------------------------------------
// prep: W_feat (A x F, f32) -> bf16
__global__ __launch_bounds__(256)
void prep_w_kernel(const float* __restrict__ w, __bf16* __restrict__ wb) {
    int i = blockIdx.x * 256 + threadIdx.x;
    wb[i] = (__bf16)w[i];
}

// prep: proj_inp[b,a] = z[b,:] . W_inp[a,:] + b_inp[a]
__global__ __launch_bounds__(256)
void prep_pi_kernel(const float* __restrict__ z, const float* __restrict__ W_inp,
                    const float* __restrict__ b_inp, float* __restrict__ pinp) {
    const int b = blockIdx.x, a = threadIdx.x;
    const float4* zp = reinterpret_cast<const float4*>(z + b * I_);
    const float4* wp = reinterpret_cast<const float4*>(W_inp + a * I_);
    float acc = b_inp[a];
#pragma unroll 4
    for (int i = 0; i < I_ / 4; ++i) {
        float4 zv = zp[i], wv = wp[i];
        acc += zv.x * wv.x + zv.y * wv.y + zv.z * wv.z + zv.w * wv.w;
    }
    pinp[b * A_ + a] = acc;
}

// ---------------------------------------------------------------------------
// scores[b,l] = sum_a relu( (features[b]^T @ W_feat^T)[l,a] + pinp[b,a] ) * v[a]
// WMMA bf16 GEMM with fused relu/dot epilogue; proj_feat never materialized.
__global__ __launch_bounds__(256)
void score_kernel(const float* __restrict__ features,
                  const __bf16* __restrict__ wfb,
                  const float* __restrict__ pinp,
                  const float* __restrict__ vat,
                  float* __restrict__ scores) {
    __shared__ __bf16 lds[LT * LDK];   // 34,816 B: features^T tile, [l][k] bf16

    const int b    = blockIdx.y;
    const int l0   = blockIdx.x * LT;
    const int t    = threadIdx.x;
    const int w    = t >> 5;
    const int lane = t & 31;
    const int m    = lane & 15;
    const int hi   = lane >> 4;

    const float* featb = features + (size_t)b * F_ * L_;

    v8f cacc[16];   // all 16 N-tiles (A = 256) kept live across K chunks
#pragma unroll
    for (int nt = 0; nt < 16; ++nt)
#pragma unroll
        for (int r = 0; r < 8; ++r) cacc[nt][r] = 0.f;

    const int sl  = t & 127;          // staging: this thread's l row
    const int sf0 = (t >> 7) * 64;    // staging: f sub-range base (0 or 64)
    const int lrow = w * 16 + m;      // this lane's l row for A-fragments

    for (int kc = 0; kc < F_ / KC; ++kc) {
        __syncthreads();
        // Stage [LT x KC] of features^T into LDS, f32 -> bf16, transposing.
        // Global reads coalesced along l; LDS b16 stores ~2-way conflicted.
#pragma unroll 8
        for (int j = 0; j < 64; ++j) {
            const int fl = sf0 + j;
            float x = featb[(size_t)(kc * KC + fl) * L_ + (l0 + sl)];
            lds[sl * LDK + fl] = (__bf16)x;
        }
        __syncthreads();

#pragma unroll
        for (int kt = 0; kt < KC / 32; ++kt) {
            // A fragment (16x32 bf16): lane m, elems 0-15 at 16 consecutive
            // k = kt*32+16*hi (matches B's contiguous order).
            const int ka = kt * 32 + hi * 16;
            bf16x8 alo = *reinterpret_cast<const bf16x8*>(&lds[lrow * LDK + ka]);
            bf16x8 ahi = *reinterpret_cast<const bf16x8*>(&lds[lrow * LDK + ka + 8]);
            v16bf a = __builtin_shufflevector(alo, ahi,
                0, 1, 2, 3, 4, 5, 6, 7, 8, 9, 10, 11, 12, 13, 14, 15);
#pragma unroll
            for (int nt = 0; nt < 16; ++nt) {
                // B fragment (32x16 bf16): lane n = nt*16+m holds 16
                // consecutive k at kt*32 + 16*hi -> two 16B loads.
                const int n = nt * 16 + m;
                const __bf16* bp = wfb + (size_t)n * F_ + kc * KC + kt * 32 + hi * 16;
                bf16x8 blo = *reinterpret_cast<const bf16x8*>(bp);
                bf16x8 bhi = *reinterpret_cast<const bf16x8*>(bp + 8);
                v16bf bb = __builtin_shufflevector(blo, bhi,
                    0, 1, 2, 3, 4, 5, 6, 7, 8, 9, 10, 11, 12, 13, 14, 15);
                cacc[nt] = __builtin_amdgcn_wmma_f32_16x16x32_bf16(
                    false, a, false, bb, (short)0, cacc[nt], false, false);
            }
        }
    }

    // Fused epilogue: relu(c + proj_inp) * v, partial per lane.
    float sacc[8];
#pragma unroll
    for (int r = 0; r < 8; ++r) sacc[r] = 0.f;
#pragma unroll
    for (int nt = 0; nt < 16; ++nt) {
        const int n  = nt * 16 + m;
        const float pi = pinp[b * A_ + n];
        const float vv = vat[n];
#pragma unroll
        for (int r = 0; r < 8; ++r) {
            float x = cacc[nt][r] + pi;
            x = x > 0.f ? x : 0.f;
            sacc[r] += x * vv;
        }
    }
    // Reduce over the 16 lanes sharing each row; C VGPR r holds row r
    // (lanes 0-15) and row r+8 (lanes 16-31).
#pragma unroll
    for (int r = 0; r < 8; ++r) {
        float s = sacc[r];
        s += __shfl_xor(s, 1, 32);
        s += __shfl_xor(s, 2, 32);
        s += __shfl_xor(s, 4, 32);
        s += __shfl_xor(s, 8, 32);
        if (m == 0)
            scores[(size_t)b * L_ + l0 + w * 16 + hi * 8 + r] = s;
    }
}

// ---------------------------------------------------------------------------
// In-place softmax over L per batch (one block per b).
__global__ __launch_bounds__(256)
void softmax_kernel(float* __restrict__ alpha) {
    const int b = blockIdx.x;
    const int t = threadIdx.x;
    float* row = alpha + (size_t)b * L_;
    __shared__ float smx[8];
    __shared__ float ssum[8];

    float vals[32];
    float mx = -3.402823466e38f;
#pragma unroll
    for (int i = 0; i < 32; ++i) {
        vals[i] = row[t + i * 256];
        mx = fmaxf(mx, vals[i]);
    }
    for (int o = 16; o > 0; o >>= 1) mx = fmaxf(mx, __shfl_xor(mx, o, 32));
    if ((t & 31) == 0) smx[t >> 5] = mx;
    __syncthreads();
    mx = smx[0];
#pragma unroll
    for (int k = 1; k < 8; ++k) mx = fmaxf(mx, smx[k]);

    float sum = 0.f;
#pragma unroll
    for (int i = 0; i < 32; ++i) {
        vals[i] = __expf(vals[i] - mx);
        sum += vals[i];
    }
    for (int o = 16; o > 0; o >>= 1) sum += __shfl_xor(sum, o, 32);
    if ((t & 31) == 0) ssum[t >> 5] = sum;
    __syncthreads();
    sum = 0.f;
#pragma unroll
    for (int k = 0; k < 8; ++k) sum += ssum[k];

    const float inv = 1.0f / sum;
#pragma unroll
    for (int i = 0; i < 32; ++i) row[t + i * 256] = vals[i] * inv;
}

// ---------------------------------------------------------------------------
// ctx[b,f] = sum_l features[b,f,l] * alpha[b,l]   (bandwidth-bound GEMV)
__global__ __launch_bounds__(256)
void ctx_kernel(const float* __restrict__ features,
                const float* __restrict__ alpha,
                float* __restrict__ ctx) {
    const int b    = blockIdx.y;
    const int f    = blockIdx.x * 8 + (threadIdx.x >> 5);
    const int lane = threadIdx.x & 31;
    const float4* fp = reinterpret_cast<const float4*>(
        features + ((size_t)b * F_ + f) * L_);
    const float4* ap = reinterpret_cast<const float4*>(alpha + (size_t)b * L_);
    float acc = 0.f;
#pragma unroll 4
    for (int i = 0; i < L_ / 128; ++i) {
        float4 fv = fp[i * 32 + lane];
        float4 av = ap[i * 32 + lane];
        acc += fv.x * av.x + fv.y * av.y + fv.z * av.z + fv.w * av.w;
    }
    for (int o = 16; o > 0; o >>= 1) acc += __shfl_xor(acc, o, 32);
    if (lane == 0) ctx[(size_t)b * F_ + f] = acc;
}

// ---------------------------------------------------------------------------
extern "C" void kernel_launch(void* const* d_in, const int* in_sizes, int n_in,
                              void* d_out, int out_size, void* d_ws, size_t ws_size,
                              hipStream_t stream) {
    (void)in_sizes; (void)n_in; (void)out_size; (void)ws_size;
    const float* z        = (const float*)d_in[0];   // [16,512]
    const float* features = (const float*)d_in[1];   // [16,512,8192]
    const float* W_feat   = (const float*)d_in[2];   // [256,512]
    const float* W_inp    = (const float*)d_in[3];   // [256,512]
    const float* b_inp    = (const float*)d_in[4];   // [256]
    const float* v_atten  = (const float*)d_in[5];   // [256]

    float* ctx   = (float*)d_out;              // [16,512]
    float* alpha = (float*)d_out + B_ * F_;    // [16,8192]; scores -> softmax in place

    float*  pinp = (float*)d_ws;                            // 16 KB
    __bf16* wfb  = (__bf16*)((char*)d_ws + 16 * 1024);      // 256 KB bf16 W_feat

    prep_w_kernel <<<A_ * F_ / 256, 256, 0, stream>>>(W_feat, wfb);
    prep_pi_kernel<<<B_, 256, 0, stream>>>(z, W_inp, b_inp, pinp);
    score_kernel  <<<dim3(L_ / LT, B_), 256, 0, stream>>>(features, wfb, pinp,
                                                          v_atten, alpha);
    softmax_kernel<<<B_, 256, 0, stream>>>(alpha);
    ctx_kernel    <<<dim3(F_ / 8, B_), 256, 0, stream>>>(features, alpha, ctx);
}
